// MapNet_60189671686742
// MI455X (gfx1250) — compile-verified, matrix operand order checked by reading, not run on hardware
//
#include <hip/hip_runtime.h>
#include <stdint.h>

#define SCALES 6
#define ITERS  4

typedef __attribute__((ext_vector_type(16))) __bf16 v16bf;
typedef __attribute__((ext_vector_type(8)))  float  v8f;

__device__ __forceinline__ unsigned short f32_to_bf16_rne(float f) {
  unsigned u = __float_as_uint(f);
  unsigned r = u + 0x7FFFu + ((u >> 16) & 1u);
  return (unsigned short)(r >> 16);
}

// ---------------------------------------------------------------- cvt weights
__global__ void cvt_bf16_kernel(const float* __restrict__ src,
                                unsigned short* __restrict__ dst, int n) {
  int i = blockIdx.x * blockDim.x + threadIdx.x;
  if (i < n) dst[i] = f32_to_bf16_rne(src[i]);
}

// ------------------------------------------------- first Linear+ReLU (K=2)
__global__ void in_layer_kernel(const float* __restrict__ x,
                                const float* __restrict__ W1,
                                const float* __restrict__ b1,
                                unsigned short* __restrict__ hbf, int total) {
  int idx = blockIdx.x * blockDim.x + threadIdx.x;
  if (idx >= total) return;
  int n = idx >> 7, d = idx & 127;
  float h = fmaf(x[2 * n], W1[2 * d], fmaf(x[2 * n + 1], W1[2 * d + 1], b1[d]));
  hbf[idx] = f32_to_bf16_rne(fmaxf(h, 0.0f));
}

// ---------------------------------------------------------------- WMMA GEMM
// 16-bit WMMA operand layout (ISA 7.12.2): lanes 0-15 hold K={0..7,16..23},
// lanes 16-31 hold K={8..15,24..31} of each 32-K block -> two contiguous 16B
// loads per lane. B mirrors A with N<->M, so W rows load with the same helper.
// All indexing is uniform-base + 32-bit offset so the backend can use the
// saddr/voffset (scale_offset) forms instead of per-lane 64-bit address math.
__device__ __forceinline__ v16bf load_tile(const uint32_t* __restrict__ base,
                                           uint32_t dwoff, int kb, int half) {
  union { uint4 u[2]; v16bf v; } t;
  const uint4* q = (const uint4*)(base + dwoff + (uint32_t)(kb * 16 + half * 4));
  t.u[0] = q[0];
  t.u[1] = q[2];
  return t.v;
}

__device__ __forceinline__ void load_a(const uint32_t* __restrict__ featb,
                                       const int* __restrict__ v_idx,
                                       int tbase, int mrow, int half, int nrows,
                                       v16bf a[4]) {
  int ae = tbase + mrow;
  if (ae >= nrows) ae = nrows - 1;              // tail clamp, keeps EXEC full
  uint32_t arow = (uint32_t)(v_idx ? v_idx[ae] : ae);
  uint32_t dwoff = arow * 64u;
#pragma unroll
  for (int kb = 0; kb < 4; ++kb) a[kb] = load_tile(featb, dwoff, kb, half);
}

__device__ __forceinline__ void store_tile(float* __restrict__ out, const v8f& c,
                                           int tbase, int col, int half,
                                           int nrows, bool full) {
  if (full) {
#pragma unroll
    for (int r = 0; r < 8; ++r)
      out[(uint32_t)(tbase + r + 8 * half) * 128u + (uint32_t)col] = c[r];
  } else {
#pragma unroll
    for (int r = 0; r < 8; ++r) {
      int e = tbase + r + 8 * half;
      if (e < nrows) out[(uint32_t)e * 128u + (uint32_t)col] = c[r];
    }
  }
}

__device__ __forceinline__ void scatter_tile(float* __restrict__ out, const v8f& c,
                                             const uint32_t uoff[8], const int valid[8],
                                             int col, bool full) {
  if (full) {
#pragma unroll
    for (int r = 0; r < 8; ++r)
      unsafeAtomicAdd(&out[uoff[r] + (uint32_t)col], c[r]);  // global_atomic_add_f32
  } else {
#pragma unroll
    for (int r = 0; r < 8; ++r)
      if (valid[r]) unsafeAtomicAdd(&out[uoff[r] + (uint32_t)col], c[r]);
  }
}

// out[u?] (+)= feat_bf16[v?] @ W.T   32 rows per wave (2 sub-tiles share B).
__global__ __launch_bounds__(256)
void gemm32_kernel(const uint32_t* __restrict__ featb,  // bf16 rows, 64 dwords/row
                   const uint32_t* __restrict__ wb,     // bf16 [128][64 dwords]
                   float* __restrict__ out,             // f32 [N][128]
                   const int* __restrict__ u_idx,
                   const int* __restrict__ v_idx,
                   int nrows) {
  int wave = (int)((blockIdx.x * blockDim.x + threadIdx.x) >> 5);
  int t0 = wave << 5;
  if (t0 >= nrows) return;                    // wave-uniform
  int t1 = t0 + 16;
  int lane = threadIdx.x & 31;
  int half = lane >> 4;
  int mrow = lane & 15;

  bool has1  = (t1 < nrows);
  bool full0 = (t0 + 16 <= nrows);
  bool full1 = (t1 + 16 <= nrows);

  v16bf a0[4], a1[4];
  load_a(featb, v_idx, t0, mrow, half, nrows, a0);
  load_a(featb, v_idx, has1 ? t1 : t0, mrow, half, nrows, a1);

  // hoist scatter indices: 8 loads per sub-tile per lane, reused for all 8 j's
  uint32_t uoff0[8], uoff1[8];
  int val0[8], val1[8];
  if (u_idx) {
#pragma unroll
    for (int r = 0; r < 8; ++r) {
      int e0 = t0 + r + 8 * half;
      int e1 = t1 + r + 8 * half;
      val0[r] = (e0 < nrows);
      val1[r] = (e1 < nrows);
      uoff0[r] = (uint32_t)u_idx[val0[r] ? e0 : 0] * 128u;
      uoff1[r] = (uint32_t)u_idx[val1[r] ? e1 : 0] * 128u;
    }
  }

#pragma unroll
  for (int j = 0; j < 8; ++j) {
    uint32_t bdw = (uint32_t)(j * 16 + mrow) * 64u;
    // preload all 4 B tiles -> one load clause, progressive loadcnt waits
    v16bf bt[4];
#pragma unroll
    for (int kb = 0; kb < 4; ++kb) bt[kb] = load_tile(wb, bdw, kb, half);

    v8f c0 = {0.f, 0.f, 0.f, 0.f, 0.f, 0.f, 0.f, 0.f};
    v8f c1 = {0.f, 0.f, 0.f, 0.f, 0.f, 0.f, 0.f, 0.f};
#pragma unroll
    for (int kb = 0; kb < 4; ++kb) {
      c0 = __builtin_amdgcn_wmma_f32_16x16x32_bf16(false, a0[kb], false, bt[kb], (short)0, c0, false, false);
      c1 = __builtin_amdgcn_wmma_f32_16x16x32_bf16(false, a1[kb], false, bt[kb], (short)0, c1, false, false);
    }
    int col = j * 16 + mrow;
    if (u_idx) {
      scatter_tile(out, c0, uoff0, val0, col, full0);
      if (has1) scatter_tile(out, c1, uoff1, val1, col, full1);
    } else {
      store_tile(out, c0, t0, col, half, nrows, full0);
      if (has1) store_tile(out, c1, t1, col, half, nrows, full1);
    }
  }
}

// ---------------------------------------------------------------- GroupNorm
__global__ __launch_bounds__(256)
void gn_kernel(const float* __restrict__ src,
               const float* __restrict__ w,
               const float* __restrict__ b,
               const float* __restrict__ addbuf,
               int do_relu,
               float* __restrict__ dst,
               unsigned short* __restrict__ dstbf,
               float* __restrict__ resdst,
               int nnodes) {
  int node = blockIdx.x * 8 + (threadIdx.x >> 5);
  if (node >= nnodes) return;
  int lane = threadIdx.x & 31;
  uint32_t base = (uint32_t)node * 128u + (uint32_t)lane * 4u;
  float4 v = *(const float4*)(src + base);
  float s = v.x + v.y + v.z + v.w;
#pragma unroll
  for (int m = 16; m >= 1; m >>= 1) s += __shfl_xor(s, m, 32);
  float mu = s * (1.0f / 128.0f);
  float d0 = v.x - mu, d1 = v.y - mu, d2 = v.z - mu, d3 = v.w - mu;
  float q = d0 * d0 + d1 * d1 + d2 * d2 + d3 * d3;
#pragma unroll
  for (int m = 16; m >= 1; m >>= 1) q += __shfl_xor(q, m, 32);
  float rs = rsqrtf(q * (1.0f / 128.0f) + 1e-5f);
  int c = lane * 4;
  float o0 = fmaf(d0 * rs, w[c + 0], b[c + 0]);
  float o1 = fmaf(d1 * rs, w[c + 1], b[c + 1]);
  float o2 = fmaf(d2 * rs, w[c + 2], b[c + 2]);
  float o3 = fmaf(d3 * rs, w[c + 3], b[c + 3]);
  if (addbuf) {
    float4 av = *(const float4*)(addbuf + base);
    o0 += av.x; o1 += av.y; o2 += av.z; o3 += av.w;
  }
  if (do_relu) {
    o0 = fmaxf(o0, 0.f); o1 = fmaxf(o1, 0.f);
    o2 = fmaxf(o2, 0.f); o3 = fmaxf(o3, 0.f);
  }
  float4 ov = make_float4(o0, o1, o2, o3);
  *(float4*)(dst + base) = ov;
  if (dstbf) {
    uint2 pk;
    pk.x = (unsigned)f32_to_bf16_rne(o0) | ((unsigned)f32_to_bf16_rne(o1) << 16);
    pk.y = (unsigned)f32_to_bf16_rne(o2) | ((unsigned)f32_to_bf16_rne(o3) << 16);
    *(uint2*)(dstbf + base) = pk;
  }
  if (resdst) *(float4*)(resdst + base) = ov;
}

// ---------------------------------------------------------------- launcher
extern "C" void kernel_launch(void* const* d_in, const int* in_sizes, int n_in,
                              void* d_out, int out_size, void* d_ws, size_t ws_size,
                              hipStream_t stream) {
  const int N  = in_sizes[0] / 2;        // 200000
  const int E  = in_sizes[2] / SCALES;   // 200000
  const int EL = in_sizes[6];            // 50000

  const float* feats   = (const float*)d_in[0];
  const float* ctrs    = (const float*)d_in[1];
  const int* pre_u     = (const int*)d_in[2];
  const int* pre_v     = (const int*)d_in[3];
  const int* suc_u     = (const int*)d_in[4];
  const int* suc_v     = (const int*)d_in[5];
  const int* left_u    = (const int*)d_in[6];
  const int* left_v    = (const int*)d_in[7];
  const int* right_u   = (const int*)d_in[8];
  const int* right_v   = (const int*)d_in[9];
  const float* W_in1   = (const float*)d_in[10];
  const float* b_in1   = (const float*)d_in[11];
  const float* W_in2   = (const float*)d_in[12];
  const float* gn_in_w = (const float*)d_in[13];
  const float* gn_in_b = (const float*)d_in[14];
  const float* W_seg1  = (const float*)d_in[15];
  const float* b_seg1  = (const float*)d_in[16];
  const float* W_seg2  = (const float*)d_in[17];
  const float* gn_seg_w = (const float*)d_in[18];
  const float* gn_seg_b = (const float*)d_in[19];
  const float* W_ctr   = (const float*)d_in[20];
  const float* W_pre   = (const float*)d_in[21];
  const float* W_suc   = (const float*)d_in[22];
  const float* W_left  = (const float*)d_in[23];
  const float* W_right = (const float*)d_in[24];
  const float* gn_norm_w = (const float*)d_in[25];
  const float* gn_norm_b = (const float*)d_in[26];
  const float* W_ctr2  = (const float*)d_in[27];
  const float* gn_ctr2_w = (const float*)d_in[28];
  const float* gn_ctr2_b = (const float*)d_in[29];

  // ---- workspace carve (all 16B aligned) ----
  char* p = (char*)d_ws;
  size_t ND = (size_t)N * 128;
  unsigned short* featb  = (unsigned short*)p; p += ND * 2;   // bf16 feat
  float* feat32          = (float*)p;          p += ND * 4;
  float* res             = (float*)p;          p += ND * 4;   // residual / input stage
  float* temp            = (float*)p;          p += ND * 4;   // f32 accumulator
  unsigned short* wb     = (unsigned short*)p;                // bf16 weights
  const size_t MSZ = 128 * 128;
  unsigned short* wb_in2   = wb;
  unsigned short* wb_seg2  = wb + 1 * MSZ;
  unsigned short* wb_ctr   = wb + 2 * MSZ;                    // [ITERS]
  unsigned short* wb_pre   = wb + 6 * MSZ;                    // [ITERS][SCALES]
  unsigned short* wb_suc   = wb + 30 * MSZ;
  unsigned short* wb_left  = wb + 54 * MSZ;                   // [ITERS]
  unsigned short* wb_right = wb + 58 * MSZ;
  unsigned short* wb_ctr2  = wb + 62 * MSZ;

  auto cvt = [&](const float* s, unsigned short* dd, int n) {
    cvt_bf16_kernel<<<(n + 255) / 256, 256, 0, stream>>>(s, dd, n);
  };
  cvt(W_in2,  wb_in2,  (int)MSZ);
  cvt(W_seg2, wb_seg2, (int)MSZ);
  cvt(W_ctr,  wb_ctr,  (int)(ITERS * MSZ));
  cvt(W_pre,  wb_pre,  (int)(ITERS * SCALES * MSZ));
  cvt(W_suc,  wb_suc,  (int)(ITERS * SCALES * MSZ));
  cvt(W_left, wb_left, (int)(ITERS * MSZ));
  cvt(W_right, wb_right, (int)(ITERS * MSZ));
  cvt(W_ctr2, wb_ctr2, (int)(ITERS * MSZ));

  auto gemm = [&](const unsigned short* fb, const unsigned short* wmat,
                  float* out, const int* u, const int* v, int nrows) {
    int waves = (nrows + 31) / 32;
    int blocks = (waves + 7) / 8;
    gemm32_kernel<<<blocks, 256, 0, stream>>>(
        (const uint32_t*)fb, (const uint32_t*)wmat, out, u, v, nrows);
  };
  auto gn = [&](const float* s, const float* wv, const float* bv, const float* add,
                int relu, float* dst, unsigned short* dbf, float* rd) {
    gn_kernel<<<(N + 7) / 8, 256, 0, stream>>>(s, wv, bv, add, relu, dst, dbf, rd, N);
  };

  // ---- input branch ----
  int tot = N * 128;
  in_layer_kernel<<<(tot + 255) / 256, 256, 0, stream>>>(ctrs, W_in1, b_in1, featb, tot);
  gemm(featb, wb_in2, temp, nullptr, nullptr, N);
  gn(temp, gn_in_w, gn_in_b, nullptr, 0, res, nullptr, nullptr);        // stage inp in res
  in_layer_kernel<<<(tot + 255) / 256, 256, 0, stream>>>(feats, W_seg1, b_seg1, featb, tot);
  gemm(featb, wb_seg2, temp, nullptr, nullptr, N);
  gn(temp, gn_seg_w, gn_seg_b, res, 1, feat32, featb, res);             // feat=relu(inp+gn(seg))

  // ---- fuse iterations ----
  for (int i = 0; i < ITERS; ++i) {
    gemm(featb, wb_ctr + (size_t)i * MSZ, temp, nullptr, nullptr, N);   // temp = feat@Wctr.T
    for (int s = 0; s < SCALES; ++s) {
      gemm(featb, wb_pre + ((size_t)i * SCALES + s) * MSZ, temp,
           pre_u + (size_t)s * E, pre_v + (size_t)s * E, E);
      gemm(featb, wb_suc + ((size_t)i * SCALES + s) * MSZ, temp,
           suc_u + (size_t)s * E, suc_v + (size_t)s * E, E);
    }
    gemm(featb, wb_left + (size_t)i * MSZ, temp, left_u, left_v, EL);
    gemm(featb, wb_right + (size_t)i * MSZ, temp, right_u, right_v, EL);

    gn(temp, gn_norm_w + i * 128, gn_norm_b + i * 128, nullptr, 1, feat32, featb, nullptr);
    gemm(featb, wb_ctr2 + (size_t)i * MSZ, temp, nullptr, nullptr, N);

    bool last = (i == ITERS - 1);
    gn(temp, gn_ctr2_w + i * 128, gn_ctr2_b + i * 128, res, 1,
       last ? (float*)d_out : feat32,
       last ? nullptr : featb,
       last ? nullptr : res);
  }
}
